// NNAttentionHead_4836133175990
// MI455X (gfx1250) — matrix operand order repaired and checked by reading, not compiled, not run on hardware
//
#include <hip/hip_runtime.h>
#include <math.h>

#define B_  2
#define T_  512
#define C_  64
#define H_  64
#define C2_ 128

typedef _Float16 v16h __attribute__((ext_vector_type(16)));
typedef _Float16 v8h  __attribute__((ext_vector_type(8)));
typedef float    v8f  __attribute__((ext_vector_type(8)));
typedef float    v4f  __attribute__((ext_vector_type(4)));

// Branch-free erf (Abramowitz-Stegun 7.1.26, |err| <= 1.5e-7) -> exact-form GELU.
// Uses raw v_rcp_f32 / v_exp_f32 (TRANS ops) so the whole thing is ~12 VALU ops,
// no divergent branches, no IEEE-division expansion.
__device__ __forceinline__ float gelu_exact(float x) {
  const float z  = fabsf(x) * 0.70710678118654752f;     // |x|/sqrt(2)
  const float t  = __builtin_amdgcn_rcpf(1.f + 0.3275911f * z);
  float p = 1.061405429f;
  p = p * t - 1.453152027f;
  p = p * t + 1.421413741f;
  p = p * t - 0.284496736f;
  p = p * t + 0.254829592f;
  const float e   = 1.f - (p * t) * __expf(-z * z);     // erf(|x|/sqrt2)
  const float erf = copysignf(e, x);
  return 0.5f * x * (1.f + erf);
}

// ---------------------------------------------------------------------------
// Kernel 1: qp = (x+pos)@W1q + b1 ; kp = (x+pos)@W1k ; v = x@Wv + bv
// Tiny (25 MFLOP total) — one block per row, plain VALU.
// ---------------------------------------------------------------------------
__global__ __launch_bounds__(64) void proj_kernel(
    const float* __restrict__ x, const float* __restrict__ pos_emb,
    const float* __restrict__ W1, const float* __restrict__ b1,
    const float* __restrict__ Wv, const float* __restrict__ bv,
    float* __restrict__ qp, float* __restrict__ kp, float* __restrict__ vv)
{
  __shared__ float xr[C_], x1r[C_];
  const int row = blockIdx.x;          // 0..B*T-1
  const int t   = row & (T_ - 1);
  const int h   = threadIdx.x;
  float xv = x[(size_t)row * C_ + h];
  xr[h]  = xv;
  x1r[h] = xv + pos_emb[t * C_ + h];
  __syncthreads();
  float aq = 0.f, ak = 0.f, av = 0.f;
#pragma unroll 8
  for (int k = 0; k < C_; ++k) {
    float x1k = x1r[k];
    ak += x1k  * W1[k * C_ + h];            // W1k = W1[0:64]
    aq += x1k  * W1[(C_ + k) * C_ + h];     // W1q = W1[64:128]
    av += xr[k] * Wv[k * H_ + h];
  }
  qp[(size_t)row * C_ + h] = aq + b1[h];    // fold b1 into qp
  kp[(size_t)row * C_ + h] = ak;
  vv[(size_t)row * H_ + h] = av + bv[h];
}

// ---------------------------------------------------------------------------
// Kernel 2: fused pp-GEMM + gelu + W2-dot + causal online-softmax + P@V
// One block = 16 query rows; 8 waves; j processed in blocks of 32 keys.
// ---------------------------------------------------------------------------
__global__ __launch_bounds__(256) void attn_kernel(
    const float* __restrict__ pde,   // (B,T,T,2C)
    const float* __restrict__ W1,    // (4C,C): rows 128..255 = W1p
    const float* __restrict__ W2,    // (C,1)
    const float* __restrict__ b2,    // (1,)
    const float* __restrict__ qp,    // (B*T,C), b1 folded in
    const float* __restrict__ kp,    // (B*T,C)
    const float* __restrict__ vv,    // (B*T,H)
    float* __restrict__ out)         // (B,T,H)
{
  __shared__ float qb[16 * C_];
  __shared__ float kb[32 * C_];
  __shared__ alignas(16) _Float16 vt[H_ * 32];   // V transposed: [h][j]
  __shared__ float wei4[4 * 512];                // per-n-tile partial logits
  __shared__ alignas(16) _Float16 p16[16 * 32];  // exp(w - m) as f16
  __shared__ float scale_s[16];
  __shared__ float lrow[16];
  __shared__ float w2s[C_];

  const int i0     = blockIdx.x * 16;
  const int b      = blockIdx.y;
  const int tid    = threadIdx.x;
  const int lane   = tid & 31;
  const int wid    = tid >> 5;            // 0..7
  const int nt     = wid & 3;             // channel n-tile (16 cols)
  const int msbase = (wid >> 2) * 16;     // M-subtile range start
  const int lane15 = lane & 15;
  const int laneHi = lane >> 4;           // 0/1
  const int ncol   = nt * 16 + lane15;

  for (int idx = tid; idx < 16 * C_; idx += 256)
    qb[idx] = qp[(size_t)(b * T_ + i0 + (idx >> 6)) * C_ + (idx & 63)];
  if (tid < C_) w2s[tid] = W2[tid];

  // B fragments of W1p (128x64), kept in VGPRs for the whole kernel.
  // 16-bit B 32x16 layout: lanes 0-15 hold K=0..7,16..23 of col N=lane,
  // lanes 16-31 hold K=8..15,24..31.
  v16h bW[4];
#pragma unroll
  for (int s = 0; s < 4; ++s) {
    const int kbase = s * 32 + laneHi * 8;
#pragma unroll
    for (int hh = 0; hh < 16; ++hh)
      bW[s][hh] = (_Float16)W1[(size_t)(2 * C_ + kbase + hh + (hh & 8)) * C_ + ncol];
  }

  v8f acc = {};                              // P@V accumulator (waves 0..3)
  float m_run = -INFINITY, l_run = 0.f;      // per-row state (tid<16)

  const int   nj  = ((i0 + 15) >> 5) + 1;    // causal: only j <= i0+15
  const float b2v = b2[0];

  __syncthreads();
  const float w2c = w2s[ncol];               // hoisted: constant per thread

  for (int jb = 0; jb < nj; ++jb) {
    const int j0 = jb * 32;
    for (int idx = tid; idx < 32 * C_; idx += 256)
      kb[idx] = kp[(size_t)(b * T_ + j0 + (idx >> 6)) * C_ + (idx & 63)];
    for (int idx = tid; idx < H_ * 32; idx += 256) {
      const int hh = idx >> 5, jj = idx & 31;
      vt[idx] = (_Float16)vv[(size_t)(b * T_ + j0 + jj) * H_ + hh];
    }
    for (int idx = tid; idx < 4 * 512; idx += 256) wei4[idx] = 0.f;
    __syncthreads();

    // ---- Phase 1: pp = A(pde)@W1p ; h = gelu(pp+qp+kp) ; wei += h.W2 ----
#pragma unroll 1
    for (int ms = 0; ms < 16; ++ms) {
      const int msub  = msbase + ms;               // 0..31
      const int il    = msub >> 1;                 // i_local (const/subtile)
      const int jlmin = (msub & 1) << 4;
      if (j0 + jlmin > i0 + il) continue;          // fully-masked subtile
      const int m  = msub * 16 + lane15;           // this lane's A row
      const float* arow =
          pde + ((size_t)(b * T_ + i0 + (m >> 5)) * T_ + (j0 + (m & 31))) * C2_;
      __builtin_prefetch(arow + 32 * C2_, 0, 0);   // next j-block, same (i,j)
      v8f c = {};
#pragma unroll
      for (int s = 0; s < 4; ++s) {
        // 16-bit A 16x32 layout: halves 0..7 = K(base..base+7),
        // halves 8..15 = K(base+16..base+23), base = laneHi*8.
        const float* ap = arow + s * 32 + laneHi * 8;
        v4f f0 = *(const v4f*)(ap);
        v4f f1 = *(const v4f*)(ap + 4);
        v4f f2 = *(const v4f*)(ap + 16);
        v4f f3 = *(const v4f*)(ap + 20);
        v16h a;
#pragma unroll
        for (int q = 0; q < 4; ++q) {
          a[q]      = (_Float16)f0[q];
          a[q + 4]  = (_Float16)f1[q];
          a[q + 8]  = (_Float16)f2[q];
          a[q + 12] = (_Float16)f3[q];
        }
        c = __builtin_amdgcn_wmma_f32_16x16x32_f16(false, a, false, bW[s],
                                                   (short)0, c, false, false);
      }
      // gelu + dot(W2) in C-fragment layout; half-wave reduce over 16 chans
      const float qv = qb[il * C_ + ncol];         // hoisted: const per task
#pragma unroll
      for (int r = 0; r < 8; ++r) {
        const int M  = msub * 16 + r + laneHi * 8;
        const int jj = M & 31;
        float xval = c[r] + qv + kb[jj * C_ + ncol];
        float t = gelu_exact(xval) * w2c;
        t += __shfl_xor(t, 1);
        t += __shfl_xor(t, 2);
        t += __shfl_xor(t, 4);
        t += __shfl_xor(t, 8);
        if (lane15 == 0) wei4[nt * 512 + M] = t;
      }
    }
    __syncthreads();

    // ---- Phase 2: online softmax, one thread per query row ----
    if (tid < 16) {
      const int ig = i0 + tid;
      float w[32];
      float bm = -INFINITY;
#pragma unroll
      for (int j = 0; j < 32; ++j) {
        const int m = tid * 32 + j;
        float s = wei4[m] + wei4[512 + m] + wei4[1024 + m] + wei4[1536 + m];
        s = (s + b2v) * 0.125f;              // * C^-0.5
        if (j0 + j > ig) s = -INFINITY;      // causal mask
        w[j] = s;
        bm = fmaxf(bm, s);
      }
      const float nm = fmaxf(m_run, bm);
      const float sc = __expf(m_run - nm);
      float ps = 0.f;
#pragma unroll
      for (int j = 0; j < 32; ++j) {
        float p = __expf(w[j] - nm);
        ps += p;
        p16[tid * 32 + j] = (_Float16)p;
      }
      l_run = l_run * sc + ps;
      m_run = nm;
      scale_s[tid] = sc;
    }
    __syncthreads();

    // ---- Phase 3: acc = acc*scale + P(16x32) @ V(32x64), waves 0..3 ----
    if (wid < 4) {
      v16h a, bfrag;
      const int kb0 = laneHi * 8;
      v8h lo = *(const v8h*)&p16[lane15 * 32 + kb0];
      v8h hi = *(const v8h*)&p16[lane15 * 32 + kb0 + 16];
#pragma unroll
      for (int q = 0; q < 8; ++q) { a[q] = lo[q]; a[q + 8] = hi[q]; }
      const v8h* vl = (const v8h*)&vt[ncol * 32 + laneHi * 8];
      v8h v0 = vl[0], v1 = vl[2];
#pragma unroll
      for (int q = 0; q < 8; ++q) { bfrag[q] = v0[q]; bfrag[q + 8] = v1[q]; }
#pragma unroll
      for (int r = 0; r < 8; ++r) acc[r] *= scale_s[r + laneHi * 8];
      acc = __builtin_amdgcn_wmma_f32_16x16x32_f16(false, a, false, bfrag,
                                                   (short)0, acc, false, false);
    }
    __syncthreads();
  }

  // ---- Finalize: out = acc / l ----
  if (tid < 16) lrow[tid] = l_run;
  __syncthreads();
  if (wid < 4) {
#pragma unroll
    for (int r = 0; r < 8; ++r) {
      const int M = r + laneHi * 8;
      out[(size_t)(b * T_ + i0 + M) * H_ + ncol] = acc[r] * (1.f / lrow[M]);
    }
  }
}

// ---------------------------------------------------------------------------
extern "C" void kernel_launch(void* const* d_in, const int* in_sizes, int n_in,
                              void* d_out, int out_size, void* d_ws, size_t ws_size,
                              hipStream_t stream) {
  const float* x   = (const float*)d_in[0];
  const float* pe  = (const float*)d_in[1];
  const float* pde = (const float*)d_in[2];
  const float* W1  = (const float*)d_in[3];
  const float* b1  = (const float*)d_in[4];
  const float* W2  = (const float*)d_in[5];
  const float* b2  = (const float*)d_in[6];
  const float* Wv  = (const float*)d_in[7];
  const float* bv  = (const float*)d_in[8];
  float* out = (float*)d_out;

  float* qp = (float*)d_ws;            // B*T*C floats
  float* kp = qp + B_ * T_ * C_;       // B*T*C floats
  float* vv = kp + B_ * T_ * C_;       // B*T*H floats

  proj_kernel<<<dim3(B_ * T_), dim3(C_), 0, stream>>>(x, pe, W1, b1, Wv, bv,
                                                      qp, kp, vv);
  attn_kernel<<<dim3(T_ / 16, B_), dim3(256), 0, stream>>>(pde, W1, W2, b2,
                                                           qp, kp, vv, out);
}
